// Custom_GRU_55052890800247
// MI455X (gfx1250) — matrix operand order repaired and checked by reading, not compiled, not run on hardware
//
#include <hip/hip_runtime.h>
#include <hip/hip_bf16.h>

typedef __attribute__((ext_vector_type(16))) _Float16 v16h;
typedef __attribute__((ext_vector_type(8)))  _Float16 v8h;
typedef __attribute__((ext_vector_type(8)))  float    v8f;

#define B_    16
#define T_    4096
#define ENO   800
#define INO   200
#define SUBN  32
#define H_    64
#define G3H   192
#define KI_PAD 224

// workspace byte offsets
#define CBE_OFF   0          // 32*800 f16  (scaled excitatory synapses)
#define CBI_OFF   51200      // 32*224 f16  (scaled+negated inhibitory, zero padded)
#define SSUB_OFF  65536      // T*B*SUB f32, layout (T, B, SUB)

__device__ __forceinline__ float sigmoidf_(float x) { return 1.0f / (1.0f + __expf(-x)); }
__device__ __forceinline__ float tanhf_(float x)    { return 1.0f - 2.0f / (__expf(2.0f * x) + 1.0f); }

// ---------------------------------------------------------------------------
// Kernel 1: fold exp(E_scale)/-exp(I_scale) into f16 copies of C_syn_e/i
// ---------------------------------------------------------------------------
__global__ void prep_weights(const float* __restrict__ Ce, const float* __restrict__ Ci,
                             const float* __restrict__ Esc, const float* __restrict__ Isc,
                             _Float16* __restrict__ CbE, _Float16* __restrict__ CbI)
{
    int tid = blockIdx.x * blockDim.x + threadIdx.x;
    if (tid < SUBN * ENO) {
        int e = tid % ENO;
        CbE[tid] = (_Float16)(Ce[tid] * __expf(Esc[e]));
    } else {
        int j = tid - SUBN * ENO;
        if (j < SUBN * KI_PAD) {
            int s = j / KI_PAD, i = j % KI_PAD;
            float v = 0.f;
            if (i < INO) v = -Ci[s * INO + i] * __expf(Isc[i]);
            CbI[s * KI_PAD + i] = (_Float16)v;
        }
    }
}

// ---------------------------------------------------------------------------
// Kernel 2: S_sub[(t,b,s)] = Se@Ce'^T + Si@Ci'^T   (WMMA f16, f32 accum)
// One wave per 16-row M tile (rows = flattened (b,t)); 2 N tiles (32 subs).
// HBM-bound: 262 MB of activations streamed once (~11 us at 23.3 TB/s).
// ---------------------------------------------------------------------------
__global__ __launch_bounds__(128) void proj_kernel(const float* __restrict__ Se,
                                                   const float* __restrict__ Si,
                                                   const _Float16* __restrict__ CbE,
                                                   const _Float16* __restrict__ CbI,
                                                   float* __restrict__ Ssub)
{
    const int wid  = (blockIdx.x * 128 + threadIdx.x) >> 5;   // 0..4095 M-tiles
    const int lane = threadIdx.x & 31;
    const int half = lane >> 4;
    const int ln   = lane & 15;

    v8f acc0 = {};   // subunit cols 0..15
    v8f acc1 = {};   // subunit cols 16..31

    // ---- excitatory: K = 800 = 25 chunks of 32 ----
    const float* arow = Se + (size_t)(wid * 16 + ln) * ENO;
    for (int kc = 0; kc < 25; ++kc) {
        v16h a;
        const int k0 = kc * 32 + 16 * half;
        #pragma unroll
        for (int i = 0; i < 8; ++i) {
            a[i]     = (_Float16)arow[k0 + i];
            a[i + 8] = (_Float16)arow[k0 + 8 + i];
        }
        const int kb = kc * 32 + 16 * half;
        const _Float16* b0p = CbE + (size_t)ln * ENO + kb;
        const _Float16* b1p = CbE + (size_t)(16 + ln) * ENO + kb;
        v8h b0lo = *(const v8h*)b0p;  v8h b0hi = *(const v8h*)(b0p + 8);
        v8h b1lo = *(const v8h*)b1p;  v8h b1hi = *(const v8h*)(b1p + 8);
        v16h b0, b1;
        #pragma unroll
        for (int i = 0; i < 8; ++i) {
            b0[i] = b0lo[i]; b0[i + 8] = b0hi[i];
            b1[i] = b1lo[i]; b1[i + 8] = b1hi[i];
        }
        acc0 = __builtin_amdgcn_wmma_f32_16x16x32_f16(false, a, false, b0, (short)0, acc0, false, false);
        acc1 = __builtin_amdgcn_wmma_f32_16x16x32_f16(false, a, false, b1, (short)0, acc1, false, false);
    }

    // ---- inhibitory: K = 200 = 6 full chunks + 1 partial ----
    const float* irow = Si + (size_t)(wid * 16 + ln) * INO;
    for (int kc = 0; kc < 7; ++kc) {
        v16h a;
        const int k0 = kc * 32 + 16 * half;
        if (kc < 6) {
            #pragma unroll
            for (int i = 0; i < 8; ++i) {
                a[i]     = (_Float16)irow[k0 + i];
                a[i + 8] = (_Float16)irow[k0 + 8 + i];
            }
        } else {
            #pragma unroll
            for (int i = 0; i < 8; ++i) {
                int ka = k0 + i, kb2 = k0 + 8 + i;
                a[i]     = (ka  < INO) ? (_Float16)irow[ka]  : (_Float16)0.f;
                a[i + 8] = (kb2 < INO) ? (_Float16)irow[kb2] : (_Float16)0.f;
            }
        }
        const int kb = kc * 32 + 16 * half;
        const _Float16* b0p = CbI + (size_t)ln * KI_PAD + kb;
        const _Float16* b1p = CbI + (size_t)(16 + ln) * KI_PAD + kb;
        v8h b0lo = *(const v8h*)b0p;  v8h b0hi = *(const v8h*)(b0p + 8);
        v8h b1lo = *(const v8h*)b1p;  v8h b1hi = *(const v8h*)(b1p + 8);
        v16h b0, b1;
        #pragma unroll
        for (int i = 0; i < 8; ++i) {
            b0[i] = b0lo[i]; b0[i + 8] = b0hi[i];
            b1[i] = b1lo[i]; b1[i + 8] = b1hi[i];
        }
        acc0 = __builtin_amdgcn_wmma_f32_16x16x32_f16(false, a, false, b0, (short)0, acc0, false, false);
        acc1 = __builtin_amdgcn_wmma_f32_16x16x32_f16(false, a, false, b1, (short)0, acc1, false, false);
    }

    // store into (T, B, SUB) layout for streaming reads in the scan
    #pragma unroll
    for (int k = 0; k < 8; ++k) {
        int row = wid * 16 + k + 8 * half;       // flattened (b,t), b = row/T
        int b = row >> 12;
        int t = row & (T_ - 1);
        size_t base = ((size_t)t * B_ + b) * SUBN;
        Ssub[base + ln]      = acc0[k];
        Ssub[base + 16 + ln] = acc1[k];
    }
}

// ---------------------------------------------------------------------------
// Kernel 3: sequential GRU scan. One workgroup per subunit, 4 waves.
// Wave w owns hidden columns [16w,16w+16); gate tiles r/z/n = N-tiles w,w+4,w+8.
// h2h weights pinned in VGPRs (f16 B-fragments); h state in LDS (A layout).
// WMMAs ordered R(a0),Z(a0),N(a0),R(a1),Z(a1),N(a1): consecutive WMMAs are
// independent, so the f16 WMMA->WMMA RAW hazard gap is filled by real work.
// ---------------------------------------------------------------------------
__global__ __launch_bounds__(128) void scan_kernel(const float* __restrict__ Ssub,
                                                   const float* __restrict__ x2hW,
                                                   const float* __restrict__ x2hb,
                                                   const float* __restrict__ h2hW,
                                                   const float* __restrict__ h2hb,
                                                   const float* __restrict__ lin_,
                                                   float* __restrict__ subout)
{
    __shared__ _Float16 hbuf[B_ * H_];   // h state, f16, [m][64]
    __shared__ float outacc[B_];

    const int s    = blockIdx.x;
    const int tid  = threadIdx.x;
    const int w    = tid >> 5;
    const int lane = tid & 31;
    const int half = lane >> 4;
    const int ln   = lane & 15;
    const int j    = w * 16 + ln;        // hidden column this lane updates

    // --- load h2h weights once into registers as WMMA B fragments ---
    v16h bf[3][2];
    #pragma unroll
    for (int gs = 0; gs < 3; ++gs) {
        const float* wrow = h2hW + ((size_t)s * G3H + gs * 64 + j) * H_;
        #pragma unroll
        for (int kc = 0; kc < 2; ++kc) {
            const int kb = kc * 32 + 8 * half;
            v16h bb;
            #pragma unroll
            for (int i = 0; i < 16; ++i) bb[i] = (_Float16)wrow[kb + i + (i >= 8 ? 8 : 0)];
            bf[gs][kc] = bb;
        }
    }

    const float wx_r = x2hW[s * G3H + j], wx_z = x2hW[s * G3H + 64 + j], wx_n = x2hW[s * G3H + 128 + j];
    const float xb_r = x2hb[s * G3H + j], xb_z = x2hb[s * G3H + 64 + j], xb_n = x2hb[s * G3H + 128 + j];
    const float hb_r = h2hb[s * G3H + j], hb_z = h2hb[s * G3H + 64 + j], hb_n = h2hb[s * G3H + 128 + j];
    const float lin  = lin_[s * H_ + j];

    for (int idx = tid; idx < B_ * H_; idx += 128) hbuf[idx] = (_Float16)0.f;
    if (tid < B_) outacc[tid] = 0.f;

    float hreg[8];
    #pragma unroll
    for (int k = 0; k < 8; ++k) hreg[k] = 0.f;

    float xc[8];
    #pragma unroll
    for (int i = 0; i < 8; ++i) xc[i] = Ssub[(size_t)(8 * half + i) * SUBN + s];

    __syncthreads();

    for (int t = 0; t < T_; ++t) {
        // A fragments from LDS h (two K-chunks of 32)
        const _Float16* hrow = hbuf + ln * H_;
        v8h a00 = *(const v8h*)(hrow + 8 * half);
        v8h a01 = *(const v8h*)(hrow + 16 + 8 * half);
        v8h a10 = *(const v8h*)(hrow + 32 + 8 * half);
        v8h a11 = *(const v8h*)(hrow + 48 + 8 * half);
        v16h a0, a1;
        #pragma unroll
        for (int i = 0; i < 8; ++i) {
            a0[i] = a00[i]; a0[i + 8] = a01[i];
            a1[i] = a10[i]; a1[i + 8] = a11[i];
        }

        // prefetch next step's x scalars (hides global latency behind WMMA)
        float xn[8];
        {
            const int tn = (t + 1 < T_) ? (t + 1) : t;
            #pragma unroll
            for (int i = 0; i < 8; ++i)
                xn[i] = Ssub[((size_t)tn * B_ + 8 * half + i) * SUBN + s];
        }

        __syncthreads();   // all waves done reading old h

        v8f accR = {}, accZ = {}, accN = {};
        // K-chunk 0 for all three gates (mutually independent)...
        accR = __builtin_amdgcn_wmma_f32_16x16x32_f16(false, a0, false, bf[0][0], (short)0, accR, false, false);
        accZ = __builtin_amdgcn_wmma_f32_16x16x32_f16(false, a0, false, bf[1][0], (short)0, accZ, false, false);
        accN = __builtin_amdgcn_wmma_f32_16x16x32_f16(false, a0, false, bf[2][0], (short)0, accN, false, false);
        // ...then K-chunk 1: each dependent WMMA is 3 slots away from its producer
        accR = __builtin_amdgcn_wmma_f32_16x16x32_f16(false, a1, false, bf[0][1], (short)0, accR, false, false);
        accZ = __builtin_amdgcn_wmma_f32_16x16x32_f16(false, a1, false, bf[1][1], (short)0, accZ, false, false);
        accN = __builtin_amdgcn_wmma_f32_16x16x32_f16(false, a1, false, bf[2][1], (short)0, accN, false, false);

        #pragma unroll
        for (int k = 0; k < 8; ++k) {
            const float x = xc[k];
            const float r = sigmoidf_(x * wx_r + xb_r + accR[k] + hb_r);
            const float z = sigmoidf_(x * wx_z + xb_z + accZ[k] + hb_z);
            const float n = tanhf_(x * wx_n + xb_n + r * (accN[k] + hb_n));
            const float hnew = n + z * (hreg[k] - n);
            hreg[k] = hnew;
            hbuf[(k + 8 * half) * H_ + j] = (_Float16)hnew;

            // reduce hnew*lin over the 16 columns of this wave's half
            float p = hnew * lin;
            p += __shfl_xor(p, 1, 32);
            p += __shfl_xor(p, 2, 32);
            p += __shfl_xor(p, 4, 32);
            p += __shfl_xor(p, 8, 32);
            if (ln == 0) atomicAdd(&outacc[k + 8 * half], p);
        }

        __syncthreads();   // h written + partials accumulated

        if (tid < B_) {
            subout[((size_t)tid * T_ + t) * SUBN + s] = outacc[tid];
            outacc[tid] = 0.f;   // reset before next iter's atomics (ordered by next barrier)
        }

        #pragma unroll
        for (int i = 0; i < 8; ++i) xc[i] = xn[i];
    }
}

// ---------------------------------------------------------------------------
// Kernel 4: final[b,t] = V_o + sum_s sub_out[b,t,s]
// ---------------------------------------------------------------------------
__global__ void final_kernel(const float* __restrict__ subout, const float* __restrict__ Vo,
                             float* __restrict__ fin)
{
    int idx = blockIdx.x * blockDim.x + threadIdx.x;
    if (idx >= B_ * T_) return;
    const float* p = subout + (size_t)idx * SUBN;
    float acc = Vo[0];
    #pragma unroll
    for (int s2 = 0; s2 < SUBN; ++s2) acc += p[s2];
    fin[idx] = acc;
}

extern "C" void kernel_launch(void* const* d_in, const int* in_sizes, int n_in,
                              void* d_out, int out_size, void* d_ws, size_t ws_size,
                              hipStream_t stream)
{
    const float* Se   = (const float*)d_in[0];
    const float* Si   = (const float*)d_in[1];
    const float* Ce   = (const float*)d_in[2];
    const float* Ci   = (const float*)d_in[3];
    const float* Esc  = (const float*)d_in[4];
    const float* Isc  = (const float*)d_in[5];
    const float* x2hW = (const float*)d_in[6];
    const float* x2hb = (const float*)d_in[7];
    const float* h2hW = (const float*)d_in[8];
    const float* h2hb = (const float*)d_in[9];
    const float* lin  = (const float*)d_in[10];
    const float* Vo   = (const float*)d_in[11];

    char* ws = (char*)d_ws;
    _Float16* CbE = (_Float16*)(ws + CBE_OFF);
    _Float16* CbI = (_Float16*)(ws + CBI_OFF);
    float*    Ssub = (float*)(ws + SSUB_OFF);

    float* fin    = (float*)d_out;               // (B,T)
    float* subout = fin + (size_t)B_ * T_;       // (B,T,SUB)

    prep_weights<<<128, 256, 0, stream>>>(Ce, Ci, Esc, Isc, CbE, CbI);
    proj_kernel<<<1024, 128, 0, stream>>>(Se, Si, CbE, CbI, Ssub);
    scan_kernel<<<SUBN, 128, 0, stream>>>(Ssub, x2hW, x2hb, h2hW, h2hb, lin, subout);
    final_kernel<<<(B_ * T_ + 255) / 256, 256, 0, stream>>>(subout, Vo, fin);
}
